// AutomatonNetwork_53231824667204
// MI455X (gfx1250) — matrix operand order repaired and checked by reading, not compiled, not run on hardware
//
#include <hip/hip_runtime.h>
#include <hip/hip_bf16.h>

// ============================================================================
// Weighted-automaton recurrence, restructured for CDNA5 WMMA.
//
//   reference:  for t in 0..16383:  p += v . pvec[c_t];  v = v @ T[c_t]
//
// Block-scan factorization (B=128 blocks of L=128 chars):
//   phase 0: T (fp32, 128x512x512) -> MT (bf16, [c][n][k] transposed layout)
//   phase 1: per block b (parallel):  P_b = prod_t T[c_t]   (bf16 WMMA chains)
//                                     q_b = sum_t R_t p[c_t] (folded in)
//            Each block is split into 4 independent 128-row slabs (row space
//            of R <- R*M never mixes), each slab = 1 workgroup with R in LDS.
//   phase 2: serial 128-step combine: p += v.q_b ; v = v P_b ; out=1-exp(p)
//
// Compute: 2*512^3*16256 ~ 4.4 TFLOP bf16 -> ~1 ms at CDNA5 WMMA rates.
// Memory:  MT = 64 MB bf16 (L2-resident, reused 128x); R slabs live in LDS.
// Staging: GLOBAL_LOAD_ASYNC_TO_LDS_B128 (ASYNCcnt) — no data VGPRs, fully
//          overlapped with the WMMA stream; one s_wait_asynccnt per chunk.
// Workspace requirement: 64MB (MT) + 64MB (P) + 256KB (q) ~ 128.3 MB in d_ws.
// ============================================================================

typedef __attribute__((ext_vector_type(16))) __bf16 v16bf;
typedef __attribute__((ext_vector_type(8)))  float  v8f;
typedef __attribute__((ext_vector_type(4)))  int    v4i;

// address-space-qualified pointer types for the async LDS-DMA builtin
typedef __attribute__((address_space(1))) v4i* gv4i_p;   // global
typedef __attribute__((address_space(3))) v4i* lv4i_p;   // LDS

#define NSTATES   512
#define ALPHA     128
#define SEQLEN    16384
#define NBLK      128                    // parallel sequence blocks
#define LSTEPS    (SEQLEN / NBLK)        // 128 chars per block
#define NSLAB     4                      // row-slabs per block
#define SLABROWS  (NSTATES / NSLAB)      // 128 rows per slab workgroup
#define SLAB_STR  520                    // LDS slab row stride (halves), bank-padded
#define CH_STR    40                     // LDS chunk row stride (halves), bank-padded
#define KCHUNK    32                     // K per WMMA (16x16x32 bf16)
#define NKCH      (NSTATES / KCHUNK)     // 16 K-chunks

#define LDS_BYTES ((SLABROWS * SLAB_STR + 2 * NSTATES * CH_STR) * 2 + NSTATES * 4)

#if __has_builtin(__builtin_amdgcn_global_load_async_to_lds_b128)
#define HAVE_ASYNC_COPY 1
#else
#define HAVE_ASYNC_COPY 0
#endif

// ---------------------------------------------------------------------------
// Phase 0: convert + transpose transfer matrices to bf16, layout MT[c][n][k]
// so that phase-1 B-chunk staging is a contiguous 64B read per column n.
// ---------------------------------------------------------------------------
__global__ __launch_bounds__(256)
void convert_transpose_bf16(const float* __restrict__ T, __bf16* __restrict__ MT) {
  const int g   = blockIdx.x * 256 + threadIdx.x;   // 0 .. 128*64*512-1
  const int c   = g >> 15;                          // / (64*512)
  const int rem = g & 32767;
  const int kq  = rem >> 9;                         // 8-wide k group
  const int n   = rem & 511;                        // coalesced read dim
  const float* src = T + ((size_t)c << 18) + n;     // element (c,k,n)
  union { __bf16 h[8]; uint4 u; } tmp;
#pragma unroll
  for (int j = 0; j < 8; ++j)
    tmp.h[j] = (__bf16)src[(size_t)(kq * 8 + j) * NSTATES];
  __bf16* dst = MT + ((size_t)c << 18) + (size_t)n * NSTATES + kq * 8;
  *(uint4*)dst = tmp.u;                             // 16B aligned store
}

// ---------------------------------------------------------------------------
// Phase 1 helpers
// ---------------------------------------------------------------------------
__device__ __forceinline__ void stage_chunk(const __bf16* __restrict__ MTc,
                                            __bf16* __restrict__ dstbuf,
                                            int kc, int tid) {
  // thread tid stages column n=tid: 32 bf16 (64B) contiguous from MT[c][n][kc..kc+32)
  const __bf16* src = MTc + (size_t)tid * NSTATES + kc;
  __bf16* dst = dstbuf + (size_t)tid * CH_STR;           // 80B stride, 16B aligned
#if HAVE_ASYNC_COPY
  // CDNA5 async path: GLOBAL_LOAD_ASYNC_TO_LDS_B128 (no data VGPRs, ASYNCcnt).
  // Pointer casts perform the required addrspacecast (generic -> AS1 / AS3).
#pragma unroll
  for (int i = 0; i < 4; ++i) {
    __builtin_amdgcn_global_load_async_to_lds_b128(
        (gv4i_p)(src + i * 8), (lv4i_p)(dst + i * 8), 0, 0);
  }
#else
  const uint4* s4 = (const uint4*)src;
  uint4 x0 = s4[0], x1 = s4[1], x2 = s4[2], x3 = s4[3];
  uint4* d4 = (uint4*)dst;
  d4[0] = x0; d4[1] = x1; d4[2] = x2; d4[3] = x3;
#endif
}

__device__ __forceinline__ void wait_staging() {
#if HAVE_ASYNC_COPY
#if __has_builtin(__builtin_amdgcn_s_wait_asynccnt)
  __builtin_amdgcn_s_wait_asynccnt(0);
#else
  asm volatile("s_wait_asynccnt 0x0" ::: "memory");     // SOPP opcode 74
#endif
#endif
}

// ---------------------------------------------------------------------------
// Phase 1: per (block, slab) workgroup: chain R <- R * T[c_t] with bf16 WMMA,
// fold q accumulation, emit P_b slab (bf16) and q_b slab (f32).
// grid = (NSLAB, NBLK), block = 512 threads (16 waves), dynamic LDS.
// ---------------------------------------------------------------------------
__global__ __launch_bounds__(512, 1)
void automaton_block_chain(const int* __restrict__ conv,
                           const float* __restrict__ pvecs,   // [ALPHA][512] f32
                           const __bf16* __restrict__ MT,     // [ALPHA][n][k] bf16
                           __bf16* __restrict__ Pg,           // [NBLK][512][512] bf16
                           float* __restrict__ qg)            // [NBLK][512] f32
{
  extern __shared__ char smem[];
  __bf16* slab   = (__bf16*)smem;                        // 128 x 520 halves
  __bf16* chunkA = slab + SLABROWS * SLAB_STR;           // 512 x 40 halves
  __bf16* chunkB = chunkA + NSTATES * CH_STR;            // 512 x 40 halves
  float*  qred   = (float*)(chunkB + NSTATES * CH_STR);  // 512 f32

  const int tid   = threadIdx.x;
  const int slabI = blockIdx.x;              // 0..3
  const int blk   = blockIdx.y;              // 0..127
  const int wave  = tid >> 5;                // 0..15
  const int lane  = tid & 31;
  const int l16   = lane & 15;
  const int khalf = lane >> 4;               // 0/1: lane half for K split
  const int strip = wave >> 1;               // output rows strip*16..+16
  const int halfN = wave & 1;                // output cols halfN*256..+256
  const int rowA  = strip * 16 + l16;        // this lane's A-fragment slab row

  // ---- R slab = identity rows [slabI*128, slabI*128+128) ----
  for (int i = tid; i < SLABROWS * SLAB_STR; i += 512) slab[i] = (__bf16)0.0f;
  __syncthreads();
  if (tid < SLABROWS) slab[tid * SLAB_STR + (slabI * SLABROWS + tid)] = (__bf16)1.0f;
  __syncthreads();

  float qreg = 0.0f;                         // partial of q[row qrow], quarter qk0/128
  const int qrow = tid >> 2;
  const int qk0  = (tid & 3) * 128;

  const v8f vzero = {0.f, 0.f, 0.f, 0.f, 0.f, 0.f, 0.f, 0.f};

  for (int t = 0; t < LSTEPS; ++t) {
    const int c = conv[blk * LSTEPS + t];
    const __bf16* MTc = MT + ((size_t)c << 18);

    // warm L2 for the next step's matrix (global_prefetch_b8)
    if (t + 1 < LSTEPS) {
      const int cn = conv[blk * LSTEPS + t + 1];
      __builtin_prefetch(MT + ((size_t)cn << 18) + (size_t)tid * 512, 0, 1);
    }

    // ---- q partial with pre-update R_t: qreg += R_t[row, qk0..+128] . p_c ----
    {
      const float* pc = pvecs + (size_t)c * NSTATES + qk0;
      const __bf16* row = slab + qrow * SLAB_STR + qk0;
      float s = 0.0f;
#pragma unroll 8
      for (int k = 0; k < 128; ++k) s += (float)row[k] * pc[k];
      qreg += s;
    }

    // ---- R' = R_slab (128x512) x M_c (512x512), fp32 WMMA accumulate ----
    v8f cacc[16];
#pragma unroll
    for (int nt = 0; nt < 16; ++nt) cacc[nt] = vzero;

    stage_chunk(MTc, chunkA, 0, tid);
    wait_staging();
    __syncthreads();

    for (int kci = 0; kci < NKCH; ++kci) {
      __bf16* cur = (kci & 1) ? chunkB : chunkA;
      if (kci + 1 < NKCH) {                       // async-stage next K-chunk
        __bf16* nxt = (kci & 1) ? chunkA : chunkB;
        stage_chunk(MTc, nxt, (kci + 1) * KCHUNK, tid);
      }
      const int kc = kci * KCHUNK;

      // A fragment (16x32 bf16): lanes 0-15 -> K kc+0..15,
      //                          lanes 16-31 -> K kc+16..31
      union { v16bf v; uint4 u[2]; } af;
      const __bf16* abase = slab + rowA * SLAB_STR + kc + khalf * 16;
      af.u[0] = *(const uint4*)abase;             // K run kc+khalf*16 .. +8
      af.u[1] = *(const uint4*)(abase + 8);       // K run kc+khalf*16+8 .. +8

#pragma unroll
      for (int nt = 0; nt < 16; ++nt) {
        const int n = halfN * 256 + nt * 16 + l16;
        // B fragment (32x16 bf16): lane half selects Klocal 0..15 / 16..31
        const __bf16* bbase = cur + (size_t)n * CH_STR + khalf * 16;
        union { v16bf v; uint4 u[2]; } bfr;
        bfr.u[0] = ((const uint4*)bbase)[0];
        bfr.u[1] = ((const uint4*)bbase)[1];
        cacc[nt] = __builtin_amdgcn_wmma_f32_16x16x32_bf16(
            false, af.v, false, bfr.v, (short)0, cacc[nt],
            /*reuse_a=*/false, /*reuse_b=*/false);
      }
      wait_staging();    // staged chunk complete (ASYNCcnt==0) before publish
      __syncthreads();
    }

    // ---- write back R' (bf16) over the slab; C layout: VGPR j -> M=j+8*khalf ----
#pragma unroll
    for (int nt = 0; nt < 16; ++nt) {
      const int n = halfN * 256 + nt * 16 + l16;
#pragma unroll
      for (int j = 0; j < 8; ++j) {
        const int m = strip * 16 + khalf * 8 + j;
        slab[m * SLAB_STR + n] = (__bf16)cacc[nt][j];
      }
    }
    __syncthreads();
  }

  // ---- reduce and emit q_b slab ----
  qred[tid] = qreg;
  __syncthreads();
  if (tid < SLABROWS) {
    const float qs = qred[tid * 4 + 0] + qred[tid * 4 + 1] +
                     qred[tid * 4 + 2] + qred[tid * 4 + 3];
    qg[blk * NSTATES + slabI * SLABROWS + tid] = qs;
  }

  // ---- emit P_b slab (bf16, row-major [row][col]) ----
  {
    const int n0 = (tid & 63) * 8;             // 8-half (16B) column group
    const int r0 = (tid >> 6) * 16;            // 16-row group
    for (int r = r0; r < r0 + 16; ++r) {
      const uint4 val = *(const uint4*)(slab + r * SLAB_STR + n0);
      __bf16* dst = Pg + ((size_t)blk * NSTATES + slabI * SLABROWS + r) * NSTATES + n0;
      *(uint4*)dst = val;
    }
  }
}

// ---------------------------------------------------------------------------
// Phase 2: serial combine over 128 blocks (single workgroup, fp32 VALU):
//   p += v . q_b ; v = v @ P_b ; then p += v . finals ; out = 1 - exp(p)
// ---------------------------------------------------------------------------
__global__ __launch_bounds__(512, 1)
void automaton_combine(const float* __restrict__ start_prob,
                       const float* __restrict__ start_vec,
                       const float* __restrict__ finals,
                       const __bf16* __restrict__ Pg,
                       const float* __restrict__ qg,
                       float* __restrict__ out)
{
  __shared__ float v[NSTATES];
  __shared__ float vn[NSTATES];
  __shared__ float pp[NSTATES];
  const int tid = threadIdx.x;
  v[tid]  = start_vec[tid];
  pp[tid] = 0.0f;
  __syncthreads();

  for (int b = 0; b < NBLK; ++b) {
    pp[tid] += v[tid] * qg[b * NSTATES + tid];       // p contribution (pre-update v)
    const __bf16* Pb = Pg + (size_t)b * NSTATES * NSTATES;
    float s = 0.0f;
#pragma unroll 8
    for (int r = 0; r < NSTATES; ++r)
      s += v[r] * (float)Pb[(size_t)r * NSTATES + tid];   // coalesced over tid
    vn[tid] = s;
    __syncthreads();
    v[tid] = vn[tid];
    __syncthreads();
  }

  pp[tid] += v[tid] * finals[tid];
  __syncthreads();
  for (int off = 256; off > 0; off >>= 1) {
    if (tid < off) pp[tid] += pp[tid + off];
    __syncthreads();
  }
  if (tid == 0) out[0] = 1.0f - __expf(start_prob[0] + pp[0]);
}

// ---------------------------------------------------------------------------
// Launcher. Workspace layout in d_ws:
//   [0, 64MB)        MT  : bf16 transposed transfer matrices [c][n][k]
//   [64MB, 128MB)    Pg  : bf16 block products P_b [b][row][col]
//   [128MB, +256KB)  qg  : f32 block functionals q_b [b][row]
// ---------------------------------------------------------------------------
extern "C" void kernel_launch(void* const* d_in, const int* in_sizes, int n_in,
                              void* d_out, int out_size, void* d_ws, size_t ws_size,
                              hipStream_t stream) {
  const int*   conv   = (const int*)  d_in[0];   // [16384] int32
  const float* sprob  = (const float*)d_in[1];   // scalar
  const float* svec   = (const float*)d_in[2];   // [512]
  const float* tmats  = (const float*)d_in[3];   // [128][512][512]
  const float* pvecs  = (const float*)d_in[4];   // [128][512]
  const float* finals = (const float*)d_in[5];   // [512]
  float* out = (float*)d_out;

  const size_t mtBytes = (size_t)ALPHA * NSTATES * NSTATES * 2;
  const size_t pBytes  = (size_t)NBLK  * NSTATES * NSTATES * 2;
  __bf16* MT = (__bf16*)d_ws;
  __bf16* Pg = (__bf16*)((char*)d_ws + mtBytes);
  float*  qg = (float*) ((char*)d_ws + mtBytes + pBytes);

  // phase 0: 128*64*512 threads, 8 elements each
  convert_transpose_bf16<<<dim3((ALPHA * 64 * NSTATES) / 256), dim3(256), 0, stream>>>(
      tmats, MT);

  // phase 1: 4 slabs x 128 blocks, 512 threads, ~212 KB dynamic LDS (<= 320 KB/WGP)
  automaton_block_chain<<<dim3(NSLAB, NBLK), dim3(512), LDS_BYTES, stream>>>(
      conv, pvecs, MT, Pg, qg);

  // phase 2: single workgroup serial combine
  automaton_combine<<<dim3(1), dim3(512), 0, stream>>>(
      sprob, svec, finals, Pg, qg, out);
}
